// GeoDSTG_87995289961068
// MI455X (gfx1250) — matrix operand order repaired and checked by reading, not compiled, not run on hardware
//
#include <hip/hip_runtime.h>
#include <hip/hip_bf16.h>

// ---------------------------------------------------------------------------
// Types / WMMA helpers (CDNA5 wave32, v_wmma_f32_16x16x32_f16)
// ---------------------------------------------------------------------------
typedef _Float16 f16_t;
typedef __attribute__((ext_vector_type(16))) _Float16 v16h;
typedef __attribute__((ext_vector_type(8)))  float    v8f;
typedef int v4i_vs __attribute__((vector_size(16)));   // matches builtin param type

static constexpr int NT = 8192;   // total nodes
static constexpr int TT = 64;     // time steps
static constexpr int HH = 64;     // hidden
static constexpr int BG = 64;     // graphs

// ---- async global->LDS copy (CDNA5 GLOBAL_LOAD_ASYNC_TO_LDS_B128) ----------
__device__ inline void async_cp16(void* lds_dst, const void* gsrc) {
#if __has_builtin(__builtin_amdgcn_global_load_async_to_lds_b128)
  void* g = const_cast<void*>(gsrc);
  __builtin_amdgcn_global_load_async_to_lds_b128(
      (__attribute__((address_space(1))) v4i_vs*)g,
      (__attribute__((address_space(3))) v4i_vs*)lds_dst, 0, 0);
#else
  *(float4*)lds_dst = *(const float4*)gsrc;   // sync fallback
#endif
}

__device__ inline void async_wait() {
#if __has_builtin(__builtin_amdgcn_s_wait_asynccnt)
  __builtin_amdgcn_s_wait_asynccnt(0);
#elif __has_builtin(__builtin_amdgcn_global_load_async_to_lds_b128)
  asm volatile("s_wait_asynccnt 0x0" ::: "memory");
#endif
}

// A fragment: 16x32 f16, row-major LDS source with leading dim lda.
// ISA layout: lane L -> M = L&15 ; half=L>>4 ; VGPR i<4: K=half*8+2i ; i>=4: K=16+half*8+2(i-4)
__device__ inline v16h load_a_frag(const f16_t* base, int lda) {
  const int lane = threadIdx.x & 31;
  const int m = lane & 15;
  const int h8 = (lane >> 4) ? 8 : 0;
  v16h a;
#pragma unroll
  for (int i = 0; i < 8; ++i) {
    const int kb = (i < 4) ? (h8 + 2 * i) : (16 + h8 + 2 * (i - 4));
    a[2 * i]     = base[m * lda + kb];
    a[2 * i + 1] = base[m * lda + kb + 1];
  }
  return a;
}

// B fragment: 32x16 f16, row-major LDS source (K rows x N cols), leading dim ldb.
// ISA layout: lane L -> N = L&15 ; half=L>>4 ; VGPR j<4: K=half*8+2j ; j>=4: K=16+half*8+2(j-4)
__device__ inline v16h load_b_frag(const f16_t* base, int ldb) {
  const int lane = threadIdx.x & 31;
  const int n = lane & 15;
  const int h8 = (lane >> 4) ? 8 : 0;
  v16h b;
#pragma unroll
  for (int j = 0; j < 8; ++j) {
    const int k = (j < 4) ? (h8 + 2 * j) : (16 + h8 + 2 * (j - 4));
    b[2 * j]     = base[k * ldb + n];
    b[2 * j + 1] = base[(k + 1) * ldb + n];
  }
  return b;
}

// C/D: lane L -> n = L&15 ; rows m = (L>>4)*8 + r for r in 0..7
__device__ inline void cfrag_coords(int& n, int& mb) {
  const int lane = threadIdx.x & 31;
  n = lane & 15;
  mb = (lane >> 4) * 8;
}

// Accumulate a 16x16 output tile: A [16 x K] @ B [K x 16], K multiple of 32.
__device__ inline v8f mma_tiles(const f16_t* A, int lda, const f16_t* B, int ldb,
                                int K, v8f acc) {
  for (int k0 = 0; k0 < K; k0 += 32) {
    v16h a = load_a_frag(A + k0, lda);
    v16h b = load_b_frag(B + k0 * ldb, ldb);
    acc = __builtin_amdgcn_wmma_f32_16x16x32_f16(false, a, false, b,
                                                 (short)0, acc, false, false);
  }
  return acc;
}

__device__ inline v8f zero8() { v8f z = {0.f,0.f,0.f,0.f,0.f,0.f,0.f,0.f}; return z; }

__device__ inline float gelu_f(float x) {
  return 0.5f * x * (1.0f + erff(x * 0.70710678118654752f));
}

__device__ inline void atomicMaxF(float* addr, float val) {
  unsigned int* ua = (unsigned int*)addr;
  unsigned int old = *ua;
  while (__uint_as_float(old) < val) {
    unsigned int assumed = old;
    old = atomicCAS(ua, assumed, __float_as_uint(val));
    if (old == assumed) break;
  }
}

// ---------------------------------------------------------------------------
// weight pre-transpose + f16 convert: WT[k*O+o] = W[o*K+k]
// ---------------------------------------------------------------------------
__global__ void convT_kernel(const float* __restrict__ W, f16_t* __restrict__ WT,
                             int O, int K) {
  int i = blockIdx.x * blockDim.x + threadIdx.x;
  if (i >= O * K) return;
  int o = i / K, k = i % K;
  WT[k * O + o] = (f16_t)W[o * K + k];
}

// ---------------------------------------------------------------------------
// Fused DATM transformer block: one block per node, 256 threads (8 waves)
// ---------------------------------------------------------------------------
struct DatmP {
  const float *ln1_g, *ln1_b, *qkv_b, *out_b, *ln2_g, *ln2_b, *b1, *cw, *cb, *b2;
  const f16_t *qkvT;  // [64][192]  (k-major)
  const f16_t *outT;  // [64][64]
  const f16_t *w1T;   // [64][256]
  const f16_t *w2T;   // [256][64]
};

__device__ inline void layernorm_rows64(const f16_t* src, f16_t* dst,
                                        const float* g, const float* b, int tid) {
  if (tid < 64) {
    const f16_t* row = src + tid * 64;
    float m = 0.f;
    for (int c = 0; c < 64; ++c) m += (float)row[c];
    m *= (1.f / 64.f);
    float v = 0.f;
    for (int c = 0; c < 64; ++c) { float d = (float)row[c] - m; v += d * d; }
    v *= (1.f / 64.f);
    const float r = rsqrtf(v + 1e-5f);
    for (int c = 0; c < 64; ++c)
      dst[tid * 64 + c] = (f16_t)(((float)row[c] - m) * r * g[c] + b[c]);
  }
}

__global__ __launch_bounds__(256) void datm_kernel(float* __restrict__ h,
                                                   const float* __restrict__ hbias,
                                                   DatmP p) {
  const int node = blockIdx.x;
  const int tid = threadIdx.x;
  const int wave = tid >> 5;
  float* hn = h + (size_t)node * (TT * HH);

  __shared__ __align__(16) f16_t sX[64 * 64];     // residual tile (f16)
  __shared__ __align__(16) f16_t sA[64 * 64];     // LN out / attn concat out
  __shared__ __align__(16) f16_t sB[64 * 64];     // weight panel / attn operands
  __shared__ __align__(16) f16_t sQKV[64 * 192];  // q|k|v, also FFN scratch
  __shared__ __align__(16) f16_t sO[64 * 64];     // scores / conv out

  // 1. load tile (+ optional broadcast node bias)
  for (int i = tid; i < 4096; i += 256) {
    int c = i & 63;
    float v = hn[i];
    if (hbias) v += hbias[(size_t)node * 64 + c];
    sX[i] = (f16_t)v;
  }
  __syncthreads();

  // 2. LN1
  layernorm_rows64(sX, sA, p.ln1_g, p.ln1_b, tid);
  __syncthreads();

  // 3. QKV projections (three 64x64x64 GEMMs); async-stage weight panels
  for (int part = 0; part < 3; ++part) {
    for (int i = tid; i < 512; i += 256) {          // panel rows stride 192 f16
      int r = i >> 3, c8 = (i & 7) * 8;
      async_cp16(sB + r * 64 + c8, p.qkvT + r * 192 + part * 64 + c8);
    }
    async_wait();
    __syncthreads();
    for (int tile = wave; tile < 16; tile += 8) {
      int mt = tile >> 2, nt = tile & 3;
      v8f acc = mma_tiles(sA + mt * 16 * 64, 64, sB + nt * 16, 64, 64, zero8());
      int n, mb; cfrag_coords(n, mb);
      int col = nt * 16 + n;
      float bias = p.qkv_b[part * 64 + col];
#pragma unroll
      for (int r = 0; r < 8; ++r)
        sQKV[(mt * 16 + mb + r) * 192 + part * 64 + col] = (f16_t)(acc[r] + bias);
    }
    __syncthreads();
  }

  // 4. attention, 4 heads of dh=16 (K padded 16->32)
  for (int hd = 0; hd < 4; ++hd) {
    for (int i = tid; i < 2048; i += 256) {           // Apad 64x32 = q_head
      int t = i >> 5, d = i & 31;
      sB[i] = (d < 16) ? sQKV[t * 192 + hd * 16 + d] : (f16_t)0.f;
    }
    for (int i = tid; i < 2048; i += 256) {           // Bpad 32x64 = k_head^T
      int d = i >> 6, s = i & 63;
      sB[2048 + i] = (d < 16) ? sQKV[s * 192 + 64 + hd * 16 + d] : (f16_t)0.f;
    }
    __syncthreads();
    for (int tile = wave; tile < 16; tile += 8) {     // scores 64x64, K=32
      int mt = tile >> 2, nt = tile & 3;
      v8f acc = mma_tiles(sB + mt * 16 * 32, 32, sB + 2048 + nt * 16, 64, 32, zero8());
      int n, mb; cfrag_coords(n, mb);
#pragma unroll
      for (int r = 0; r < 8; ++r)
        sO[(mt * 16 + mb + r) * 64 + nt * 16 + n] = (f16_t)acc[r];
    }
    __syncthreads();
    if (tid < 64) {                                   // softmax row, scale 1/sqrt(16)
      f16_t* row = sO + tid * 64;
      float mx = -1e30f;
      for (int s = 0; s < 64; ++s) mx = fmaxf(mx, 0.25f * (float)row[s]);
      float sum = 0.f;
      for (int s = 0; s < 64; ++s) {
        float e = expf(0.25f * (float)row[s] - mx);
        row[s] = (f16_t)e; sum += e;
      }
      float inv = 1.f / sum;
      for (int s = 0; s < 64; ++s) row[s] = (f16_t)((float)row[s] * inv);
    }
    __syncthreads();
    for (int tile = wave; tile < 4; tile += 8) {      // attn @ V -> sA head cols
      int mt = tile;
      v8f acc = mma_tiles(sO + mt * 16 * 64, 64, sQKV + 128 + hd * 16, 192, 64, zero8());
      int n, mb; cfrag_coords(n, mb);
#pragma unroll
      for (int r = 0; r < 8; ++r)
        sA[(mt * 16 + mb + r) * 64 + hd * 16 + n] = (f16_t)acc[r];
    }
    __syncthreads();
  }

  // 5. out projection + residual -> sX
  for (int i = tid; i < 512; i += 256)
    async_cp16(sB + i * 8, p.outT + i * 8);
  async_wait();
  __syncthreads();
  for (int tile = wave; tile < 16; tile += 8) {
    int mt = tile >> 2, nt = tile & 3;
    v8f acc = mma_tiles(sA + mt * 16 * 64, 64, sB + nt * 16, 64, 64, zero8());
    int n, mb; cfrag_coords(n, mb);
    int col = nt * 16 + n;
#pragma unroll
    for (int r = 0; r < 8; ++r) {
      int row = mt * 16 + mb + r;
      float v = acc[r] + p.out_b[col] + (float)sX[row * 64 + col];
      sX[row * 64 + col] = (f16_t)v;
    }
  }
  __syncthreads();

  // 6. LN2
  layernorm_rows64(sX, sA, p.ln2_g, p.ln2_b, tid);
  __syncthreads();

  // 7. FFN with fused depthwise conv; 4 channel chunks of 64, acc in VGPRs
  f16_t* sS1 = sQKV;  // reuse (q/k/v dead now)
  v8f accF0 = zero8(), accF1 = zero8();
  for (int cc = 0; cc < 4; ++cc) {
    for (int i = tid; i < 512; i += 256) {            // w1 chunk (rows stride 256)
      int r = i >> 3, c8 = (i & 7) * 8;
      async_cp16(sB + r * 64 + c8, p.w1T + r * 256 + cc * 64 + c8);
    }
    async_wait();
    __syncthreads();
    for (int tile = wave; tile < 16; tile += 8) {     // gelu(ln2 @ w1^T + b1)
      int mt = tile >> 2, nt = tile & 3;
      v8f acc = mma_tiles(sA + mt * 16 * 64, 64, sB + nt * 16, 64, 64, zero8());
      int n, mb; cfrag_coords(n, mb);
      int col = nt * 16 + n;
#pragma unroll
      for (int r = 0; r < 8; ++r)
        sS1[(mt * 16 + mb + r) * 64 + col] = (f16_t)gelu_f(acc[r] + p.b1[cc * 64 + col]);
    }
    __syncthreads();
    for (int i = tid; i < 4096; i += 256) {           // dwconv over T + gelu
      int t = i >> 6, c = i & 63;
      int ch = cc * 64 + c;
      float a = p.cb[ch];
      float w0 = p.cw[ch * 3 + 0], w1c = p.cw[ch * 3 + 1], w2c = p.cw[ch * 3 + 2];
      if (t > 0)  a += w0  * (float)sS1[(t - 1) * 64 + c];
      a += w1c * (float)sS1[t * 64 + c];
      if (t < 63) a += w2c * (float)sS1[(t + 1) * 64 + c];
      sO[i] = (f16_t)gelu_f(a);
    }
    __syncthreads();
    for (int i = tid; i < 512; i += 256)              // w2 chunk (contiguous 8KB)
      async_cp16(sB + i * 8, p.w2T + cc * 4096 + i * 8);
    async_wait();
    __syncthreads();
    {
      int mt = wave >> 2, nt = wave & 3;
      accF0 = mma_tiles(sO + mt * 16 * 64, 64, sB + nt * 16, 64, 64, accF0);
    }
    {
      int tile = wave + 8;
      int mt = tile >> 2, nt = tile & 3;
      accF1 = mma_tiles(sO + mt * 16 * 64, 64, sB + nt * 16, 64, 64, accF1);
    }
    __syncthreads();
  }

  // 8. residual + b2 -> global f32
  {
    int n, mb; cfrag_coords(n, mb);
    {
      int mt = wave >> 2, nt = wave & 3;
      int col = nt * 16 + n;
#pragma unroll
      for (int r = 0; r < 8; ++r) {
        int row = mt * 16 + mb + r;
        hn[row * 64 + col] = accF0[r] + p.b2[col] + (float)sX[row * 64 + col];
      }
    }
    {
      int tile = wave + 8;
      int mt = tile >> 2, nt = tile & 3;
      int col = nt * 16 + n;
#pragma unroll
      for (int r = 0; r < 8; ++r) {
        int row = mt * 16 + mb + r;
        hn[row * 64 + col] = accF1[r] + p.b2[col] + (float)sX[row * 64 + col];
      }
    }
  }
}

// ---------------------------------------------------------------------------
// Generic row GEMM: out[M,NC] = act(A[M,64] @ W[NC,64]^T + bias), WMMA tiles
// ---------------------------------------------------------------------------
static constexpr int RG_RELU = 1, RG_ACC = 2, RG_ROWSCALE = 4;

__global__ __launch_bounds__(256) void rowgemm_kernel(
    const float* __restrict__ A, const float* __restrict__ W,
    const f16_t* __restrict__ W16, const float* __restrict__ bias,
    float* __restrict__ out, f16_t* __restrict__ out16,
    const float* __restrict__ rowscale, int M, int NC, int flags) {
  __shared__ __align__(16) f16_t sA[64 * 64];
  __shared__ __align__(16) f16_t sB[64 * 64];
  const int tid = threadIdx.x, wave = tid >> 5;
  const int row0 = blockIdx.x * 64;
  for (int i = tid; i < 4096; i += 256) {
    int r = i >> 6, c = i & 63;
    int row = row0 + r;
    float v = (row < M) ? A[(size_t)row * 64 + c] : 0.f;
    if (flags & RG_ROWSCALE) {
      float cntv = (row < M) ? rowscale[row] : 1.f;
      v *= 1.f / fmaxf(cntv, 1.f);
    }
    sA[i] = (f16_t)v;
  }
  if (W16 && NC == 64) {                      // pre-converted panel: async copy
    for (int i = tid; i < 512; i += 256)
      async_cp16(sB + i * 8, W16 + i * 8);
    async_wait();
  } else {
    for (int i = tid; i < 4096; i += 256) {
      int k = i >> 6, n = i & 63;
      sB[i] = (n < NC) ? (f16_t)W[n * 64 + k] : (f16_t)0.f;
    }
  }
  __syncthreads();
  const int ntN = (NC + 15) >> 4;
  const int ntiles = 4 * ntN;
  for (int tile = wave; tile < ntiles; tile += 8) {
    int mt = tile / ntN, nt = tile % ntN;
    v8f acc = mma_tiles(sA + mt * 16 * 64, 64, sB + nt * 16, 64, 64, zero8());
    int n, mb; cfrag_coords(n, mb);
    int col = nt * 16 + n;
    if (col < NC) {
#pragma unroll
      for (int r = 0; r < 8; ++r) {
        int row = row0 + mt * 16 + mb + r;
        if (row < M) {
          float v = acc[r] + (bias ? bias[col] : 0.f);
          if (flags & RG_RELU) v = fmaxf(v, 0.f);
          size_t oi = (size_t)row * NC + col;
          if (flags & RG_ACC) out[oi] += v; else out[oi] = v;
          if (out16) out16[oi] = (f16_t)v;
        }
      }
    }
  }
}

// ---------------------------------------------------------------------------
// adj mixing: per graph b, hd[b] = adj[128,128] @ y[b,128,64]   (WMMA, K=128)
// ---------------------------------------------------------------------------
__global__ __launch_bounds__(256) void adjmm_kernel(const f16_t* __restrict__ adj16,
                                                    const f16_t* __restrict__ y16,
                                                    float* __restrict__ hd) {
  __shared__ __align__(16) f16_t sAdj[128 * 128];
  __shared__ __align__(16) f16_t sY[128 * 64];
  const int b = blockIdx.x, tid = threadIdx.x, wave = tid >> 5;
  for (int i = tid; i < 2048; i += 256)
    async_cp16(sAdj + i * 8, adj16 + i * 8);
  for (int i = tid; i < 1024; i += 256)
    async_cp16(sY + i * 8, y16 + (size_t)b * 8192 + i * 8);
  async_wait();
  __syncthreads();
  for (int tile = wave; tile < 32; tile += 8) {
    int mt = tile >> 2, nt = tile & 3;
    v8f acc = mma_tiles(sAdj + mt * 16 * 128, 128, sY + nt * 16, 64, 128, zero8());
    int n, mb; cfrag_coords(n, mb);
    int col = nt * 16 + n;
#pragma unroll
    for (int r = 0; r < 8; ++r) {
      int row = mt * 16 + mb + r;
      hd[((size_t)b * 128 + row) * 64 + col] = acc[r];
    }
  }
}

// ---------------------------------------------------------------------------
// small scalar kernels
// ---------------------------------------------------------------------------
__global__ void inproj_kernel(const float* __restrict__ x, const float* __restrict__ inw,
                              const float* __restrict__ inb, const float* __restrict__ pt,
                              float* __restrict__ h) {
  long idx = (long)blockIdx.x * blockDim.x + threadIdx.x;
  if (idx >= (long)NT * TT * HH) return;
  int c = (int)(idx & 63);
  long nt = idx >> 6;
  int t = (int)(nt & 63);
  long n = nt >> 6;
  h[idx] = x[n * 64 + t] * inw[c] + inb[c] + pt[t * 64 + c];
}

__global__ void mean_kernel(const float* __restrict__ h, const float* __restrict__ posnode,
                            float* __restrict__ xs) {
  int i = blockIdx.x * blockDim.x + threadIdx.x;
  if (i >= NT * HH) return;
  int n = i >> 6, c = i & 63;
  float s = 0.f;
  for (int t = 0; t < 64; ++t) s += h[(size_t)n * 4096 + t * 64 + c];
  float v = s * (1.f / 64.f);
  if (posnode) v += posnode[(n & 127) * 64 + c];
  xs[i] = v;
}

__global__ void adjgen_kernel(const float* __restrict__ w1, const float* __restrict__ w2,
                              f16_t* __restrict__ adj16) {
  __shared__ float red[128];
  int i = blockIdx.x, j = threadIdx.x;
  float dot = 0.f;
  for (int k = 0; k < 64; ++k) dot += w1[i * 64 + k] * w2[j * 64 + k];
  float v = fmaxf(dot, 0.f) * 3.0f;
  red[j] = v; __syncthreads();
  for (int s = 64; s > 0; s >>= 1) { if (j < s) red[j] = fmaxf(red[j], red[j + s]); __syncthreads(); }
  float mx = red[0]; __syncthreads();
  float e = expf(v - mx);
  red[j] = e; __syncthreads();
  for (int s = 64; s > 0; s >>= 1) { if (j < s) red[j] += red[j + s]; __syncthreads(); }
  adj16[i * 128 + j] = (f16_t)(e / red[0]);
}

__global__ void seg_init_kernel(float* smax, float* den, float* cnt, float* agg) {
  int i = blockIdx.x * blockDim.x + threadIdx.x;
  if (i >= NT * 64) return;
  agg[i] = 0.f;
  if ((i & 63) == 0) { int n = i >> 6; smax[n] = -1e30f; den[n] = 0.f; cnt[n] = 0.f; }
}

__global__ void edge_score_kernel(const int* __restrict__ src, const int* __restrict__ dst,
                                  const float* __restrict__ q, const float* __restrict__ k,
                                  float* __restrict__ es, float* __restrict__ smax,
                                  const float* __restrict__ temp, int E, int F) {
  int e = blockIdx.x * blockDim.x + threadIdx.x;
  if (e >= F) return;
  int s, d;
  if (e < E) { s = src[e]; d = dst[e]; } else { s = d = e - E; }
  float dot = 0.f;
  for (int c = 0; c < 64; ++c) dot += q[(size_t)d * 64 + c] * k[(size_t)s * 64 + c];
  float sc = dot * 0.125f / temp[0];
  es[e] = sc;
  atomicMaxF(smax + d, sc);
}

__global__ void edge_exp_kernel(const int* __restrict__ dst, float* __restrict__ es,
                                const float* __restrict__ smax, float* __restrict__ den,
                                float* __restrict__ cnt, int E, int F) {
  int e = blockIdx.x * blockDim.x + threadIdx.x;
  if (e >= F) return;
  int d = (e < E) ? dst[e] : e - E;
  float num = expf(es[e] - smax[d]);
  es[e] = num;
  atomicAdd(den + d, num);
  atomicAdd(cnt + d, 1.0f);
}

__global__ void edge_scatter_kernel(const int* __restrict__ src, const int* __restrict__ dst,
                                    const float* __restrict__ ew, const float* __restrict__ es,
                                    const float* __restrict__ den, const float* __restrict__ xf,
                                    float* __restrict__ agg, int E, int F) {
  long idx = (long)blockIdx.x * blockDim.x + threadIdx.x;
  int e = (int)(idx >> 6);
  int c = (int)(idx & 63);
  if (e >= F) return;
  int s, d; float w;
  if (e < E) { s = src[e]; d = dst[e]; w = ew[e]; } else { s = d = e - E; w = 1.f; }
  float coef = es[e] / den[d] * w;
  atomicAdd(agg + (size_t)d * 64 + c, xf[(size_t)s * 64 + c] * coef);
}

__global__ void gate_ln_kernel(const float* __restrict__ hs, const float* __restrict__ hd,
                               const float* __restrict__ resid, const float* __restrict__ gw,
                               const float* __restrict__ gb, const float* __restrict__ lg,
                               const float* __restrict__ lb, float* __restrict__ out) {
  int i = blockIdx.x * blockDim.x + threadIdx.x;
  if (i >= NT) return;
  const float* a = hs + (size_t)i * 64;
  const float* b = hd + (size_t)i * 64;
  float zacc = gb[0];
  for (int c = 0; c < 64; ++c) zacc += a[c] * gw[c] + b[c] * gw[64 + c];
  float z = 1.f / (1.f + expf(-zacc));
  float hf[64];
  float m = 0.f;
  for (int c = 0; c < 64; ++c) {
    hf[c] = z * a[c] + (1.f - z) * b[c] + resid[(size_t)i * 64 + c];
    m += hf[c];
  }
  m *= (1.f / 64.f);
  float v = 0.f;
  for (int c = 0; c < 64; ++c) { float d = hf[c] - m; v += d * d; }
  v *= (1.f / 64.f);
  float r = rsqrtf(v + 1e-5f);
  for (int c = 0; c < 64; ++c) out[(size_t)i * 64 + c] = (hf[c] - m) * r * lg[c] + lb[c];
}

__global__ void pool_kernel(const float* __restrict__ fin, const float* __restrict__ sw,
                            const float* __restrict__ sb, float* __restrict__ gr) {
  __shared__ float red[128];
  __shared__ float aw[128];
  int b = blockIdx.x, t = threadIdx.x;   // 128 threads
  const float* fb = fin + (size_t)b * 128 * 64;
  float dot = sb[0];
  for (int c = 0; c < 64; ++c) dot += fb[t * 64 + c] * sw[c];
  float a = tanhf(dot);
  red[t] = a; __syncthreads();
  for (int s = 64; s > 0; s >>= 1) { if (t < s) red[t] = fmaxf(red[t], red[t + s]); __syncthreads(); }
  float mx = red[0]; __syncthreads();
  float e = expf(a - mx);
  red[t] = e; __syncthreads();
  for (int s = 64; s > 0; s >>= 1) { if (t < s) red[t] += red[t + s]; __syncthreads(); }
  float inv = 1.f / red[0];
  aw[t] = e * inv;
  __syncthreads();
  if (t < 64) {
    float g = 0.f;
    for (int n = 0; n < 128; ++n) g += fb[n * 64 + t] * aw[n];
    gr[b * 64 + t] = g;
  }
}

__global__ void fc2_kernel(const float* __restrict__ tmp, const float* __restrict__ w,
                           const float* __restrict__ b, float* __restrict__ out) {
  int i = blockIdx.x * blockDim.x + threadIdx.x;
  if (i >= 64) return;
  float acc = b[0];
  for (int k = 0; k < 32; ++k) acc += tmp[i * 32 + k] * w[k];
  out[i] = acc;
}

// ---------------------------------------------------------------------------
// Host launcher
// ---------------------------------------------------------------------------
extern "C" void kernel_launch(void* const* d_in, const int* in_sizes, int n_in,
                              void* d_out, int out_size, void* d_ws, size_t ws_size,
                              hipStream_t stream) {
  (void)n_in; (void)out_size; (void)ws_size;
  auto P = [&](int i) { return (const float*)d_in[i]; };
  const float* x = P(0);
  const int* eidx = (const int*)d_in[1];
  const float* ew = P(2);
  const int E = in_sizes[2];
  const int F = E + NT;
  const int* esrc = eidx;
  const int* edst = eidx + E;

  // ---- workspace carve: f32 region then f16 region ----
  float* W = (float*)d_ws;
  size_t off = 0;
  auto alloc = [&](size_t n) { float* p = W + off; off += n; return p; };
  float* h    = alloc((size_t)NT * TT * HH);  // 33.5M floats
  float* xs   = alloc((size_t)NT * HH);
  float* qb   = alloc((size_t)NT * HH);
  float* kb   = alloc((size_t)NT * HH);
  float* agg  = alloc((size_t)NT * HH);
  float* hs   = alloc((size_t)NT * HH);
  float* yd   = alloc((size_t)NT * HH);
  float* hd   = alloc((size_t)NT * HH);
  float* hs1  = alloc((size_t)NT * HH);
  float* es   = alloc((size_t)F);
  float* smax = alloc(NT);
  float* den  = alloc(NT);
  float* cnt  = alloc(NT);
  float* gr   = alloc(BG * HH);
  float* tmp  = alloc(BG * 32);

  f16_t* H16 = (f16_t*)(W + off);
  size_t off16 = 0;
  auto alloc16 = [&](size_t n) { f16_t* p = H16 + off16; off16 += n; return p; };
  f16_t* qkvT1 = alloc16(192 * 64);
  f16_t* outT1 = alloc16(64 * 64);
  f16_t* w1T1  = alloc16(64 * 256);
  f16_t* w2T1  = alloc16(256 * 64);
  f16_t* qkvT2 = alloc16(192 * 64);
  f16_t* outT2 = alloc16(64 * 64);
  f16_t* w1T2  = alloc16(64 * 256);
  f16_t* w2T2  = alloc16(256 * 64);
  f16_t* ws1c  = alloc16(64 * 64);
  f16_t* wn1c  = alloc16(64 * 64);
  f16_t* wq1c  = alloc16(64 * 64);
  f16_t* wk1c  = alloc16(64 * 64);
  f16_t* ws2c  = alloc16(64 * 64);
  f16_t* wn2c  = alloc16(64 * 64);
  f16_t* wq2c  = alloc16(64 * 64);
  f16_t* wk2c  = alloc16(64 * 64);
  f16_t* wd1c  = alloc16(64 * 64);
  f16_t* wd2c  = alloc16(64 * 64);
  f16_t* adj16 = alloc16(128 * 128);
  f16_t* yd16  = alloc16((size_t)NT * HH);

  const int TB = 256;
  auto conv = [&](const float* src, f16_t* dst, int O, int K) {
    convT_kernel<<<(O * K + TB - 1) / TB, TB, 0, stream>>>(src, dst, O, K);
  };
  // weight pre-transpose/convert (one-time, small)
  conv(P(11), qkvT1, 192, 64); conv(P(13), outT1, 64, 64);
  conv(P(17), w1T1, 256, 64);  conv(P(21), w2T1, 64, 256);
  conv(P(25), qkvT2, 192, 64); conv(P(27), outT2, 64, 64);
  conv(P(31), w1T2, 256, 64);  conv(P(35), w2T2, 64, 256);
  conv(P(37), ws1c, 64, 64);   conv(P(39), wn1c, 64, 64);
  conv(P(41), wq1c, 64, 64);   conv(P(43), wk1c, 64, 64);
  conv(P(46), ws2c, 64, 64);   conv(P(48), wn2c, 64, 64);
  conv(P(50), wq2c, 64, 64);   conv(P(52), wk2c, 64, 64);
  conv(P(55), wd1c, 64, 64);   conv(P(57), wd2c, 64, 64);

  DatmP dp1 = { P(9), P(10), P(12), P(14), P(15), P(16), P(18), P(19), P(20), P(22),
                qkvT1, outT1, w1T1, w2T1 };
  DatmP dp2 = { P(23), P(24), P(26), P(28), P(29), P(30), P(32), P(33), P(34), P(36),
                qkvT2, outT2, w1T2, w2T2 };

  // input projection + temporal pos
  {
    long tot = (long)NT * TT * HH;
    inproj_kernel<<<(unsigned)((tot + TB - 1) / TB), TB, 0, stream>>>(x, P(3), P(4), P(5), h);
  }
  // DATM1
  datm_kernel<<<NT, TB, 0, stream>>>(h, nullptr, dp1);
  // temporal mean + node pos
  mean_kernel<<<(NT * HH + TB - 1) / TB, TB, 0, stream>>>(h, P(6), xs);
  // learned adjacency (writes f16 directly)
  adjgen_kernel<<<128, 128, 0, stream>>>(P(7), P(8), adj16);

  auto run_gagb = [&](int base, const float* xf, const f16_t* wq16, const f16_t* wk16,
                      const f16_t* wself16, const f16_t* wneigh16) {
    rowgemm_kernel<<<NT / 64, TB, 0, stream>>>(xf, P(base + 4), wq16, P(base + 5),
                                               qb, nullptr, nullptr, NT, 64, 0);
    rowgemm_kernel<<<NT / 64, TB, 0, stream>>>(xf, P(base + 6), wk16, P(base + 7),
                                               kb, nullptr, nullptr, NT, 64, 0);
    seg_init_kernel<<<(NT * 64 + TB - 1) / TB, TB, 0, stream>>>(smax, den, cnt, agg);
    edge_score_kernel<<<(F + TB - 1) / TB, TB, 0, stream>>>(esrc, edst, qb, kb, es, smax,
                                                            P(base + 8), E, F);
    edge_exp_kernel<<<(F + TB - 1) / TB, TB, 0, stream>>>(edst, es, smax, den, cnt, E, F);
    {
      long tot = (long)F * 64;
      edge_scatter_kernel<<<(unsigned)((tot + TB - 1) / TB), TB, 0, stream>>>(
          esrc, edst, ew, es, den, xf, agg, E, F);
    }
    rowgemm_kernel<<<NT / 64, TB, 0, stream>>>(xf, P(base + 0), wself16, P(base + 1),
                                               hs, nullptr, nullptr, NT, 64, 0);
    rowgemm_kernel<<<NT / 64, TB, 0, stream>>>(agg, P(base + 2), wneigh16, P(base + 3),
                                               hs, nullptr, cnt, NT, 64, RG_ACC | RG_ROWSCALE);
  };

  // round 1: gagb1 + dyn1 + gate1 + ns1
  run_gagb(37, xs, wq1c, wk1c, ws1c, wn1c);
  rowgemm_kernel<<<NT / 64, TB, 0, stream>>>(xs, P(55), wd1c, P(56), yd, yd16,
                                             nullptr, NT, 64, 0);
  adjmm_kernel<<<BG, TB, 0, stream>>>(adj16, yd16, hd);
  gate_ln_kernel<<<(NT + TB - 1) / TB, TB, 0, stream>>>(hs, hd, xs, P(59), P(60),
                                                        P(63), P(64), hs1);

  // DATM2 (input = h + hs1 broadcast over T), in-place
  datm_kernel<<<NT, TB, 0, stream>>>(h, hs1, dp2);
  mean_kernel<<<(NT * HH + TB - 1) / TB, TB, 0, stream>>>(h, nullptr, xs);

  // round 2: gagb2 + dyn2 + gate2 + ns2 -> fin (reuse hs1)
  run_gagb(46, xs, wq2c, wk2c, ws2c, wn2c);
  rowgemm_kernel<<<NT / 64, TB, 0, stream>>>(xs, P(57), wd2c, P(58), yd, yd16,
                                             nullptr, NT, 64, 0);
  adjmm_kernel<<<BG, TB, 0, stream>>>(adj16, yd16, hd);
  gate_ln_kernel<<<(NT + TB - 1) / TB, TB, 0, stream>>>(hs, hd, xs, P(61), P(62),
                                                        P(65), P(66), hs1);

  // sensor-attention pooling + head
  pool_kernel<<<BG, 128, 0, stream>>>(hs1, P(67), P(68), gr);
  rowgemm_kernel<<<1, TB, 0, stream>>>(gr, P(69), nullptr, P(70), tmp, nullptr,
                                       nullptr, BG, 32, RG_RELU);
  fc2_kernel<<<1, 64, 0, stream>>>(tmp, P(71), P(72), (float*)d_out);
}
